// QuantumPolicy_56135222558892
// MI455X (gfx1250) — matrix operand order repaired and checked by reading, not compiled, not run on hardware
//
#include <hip/hip_runtime.h>
#include <hip/hip_bf16.h>
#include <math.h>

typedef _Float16 v16h __attribute__((ext_vector_type(16)));
typedef _Float16 v8h  __attribute__((ext_vector_type(8)));
typedef float    v8f  __attribute__((ext_vector_type(8)));

#define B_TOTAL 262144
#define S_DIM   64
#define H1_DIM  256
#define H2_DIM  128

#define WAVES 2
#define ROW_TILES 2                       // 16-row WMMA tiles per wave
#define ROWS_PER_WAVE (ROW_TILES * 16)    // 32
#define ROWS_PER_BLOCK (WAVES * ROWS_PER_WAVE)  // 64

// Packed-weight region sizes (in _Float16 elements). Each WMMA B-tile is
// 32 lanes x 16 halfs = 512 halfs, stored so lane l loads a contiguous v16h
// at tile*512 + l*16.
#define PW1_TILES (2 * 16)   // K-steps(64/32) x N-tiles(256/16)
#define PW2_TILES (8 * 8)    // K-steps(256/32) x N-tiles(128/16)
#define PW3_TILES (4 * 1)    // K-steps(128/32) x 1 padded N-tile
#define PW1_HALFS (PW1_TILES * 512)
#define PW2_HALFS (PW2_TILES * 512)
#define PW3_HALFS (PW3_TILES * 512)
#define PACK_TOTAL (PW1_HALFS + PW2_HALFS + PW3_HALFS)
// G matrix (16 floats) lives right after the packed halfs (offset is 16B aligned).
#define G_BYTE_OFFSET (PACK_TOTAL * 2)

// ---------------------------------------------------------------------------
// Pre-pack fp32 weights into f16 WMMA B-matrix register layout.
// B-layout (16-bit, 32x16 per tile): lane l holds column n = 16*j + (l&15);
// lanes 0-15 hold K = kk*32 + [0..7],[16..23], lanes 16-31 hold K = kk*32 + [8..15],[24..31].
// ---------------------------------------------------------------------------
__global__ void pack_weights_kernel(const float* __restrict__ W1,
                                    const float* __restrict__ W2,
                                    const float* __restrict__ W3,
                                    _Float16* __restrict__ packed) {
  int idx = blockIdx.x * blockDim.x + threadIdx.x;
  if (idx >= PACK_TOTAL) return;
  float val;
  if (idx < PW1_HALFS) {
    int t = idx >> 9, r = idx & 511;
    int l = r >> 4, hv = r & 15;
    int kk = t >> 4, j = t & 15;
    int k = kk * 32 + ((l < 16) ? 0 : 8) + hv + ((hv < 8) ? 0 : 8);
    int n = j * 16 + (l & 15);
    val = W1[k * H1_DIM + n];
  } else if (idx < PW1_HALFS + PW2_HALFS) {
    int i2 = idx - PW1_HALFS;
    int t = i2 >> 9, r = i2 & 511;
    int l = r >> 4, hv = r & 15;
    int kk = t >> 3, j = t & 7;
    int k = kk * 32 + ((l < 16) ? 0 : 8) + hv + ((hv < 8) ? 0 : 8);
    int n = j * 16 + (l & 15);
    val = W2[k * H2_DIM + n];
  } else {
    int i3 = idx - PW1_HALFS - PW2_HALFS;
    int t = i3 >> 9, r = i3 & 511;
    int l = r >> 4, hv = r & 15;
    int k = t * 32 + ((l < 16) ? 0 : 8) + hv + ((hv < 8) ? 0 : 8);
    int n = l & 15;
    val = (n < 2) ? W3[k * 2 + n] : 0.0f;   // pad N: 2 -> 16 with zeros
  }
  packed[idx] = (_Float16)val;
}

// ---------------------------------------------------------------------------
// Collapse the batch-independent part of the VQC into a real symmetric 4x4
// matrix G so that per row: z = r^T G r with r = (c0c1, c0s1, s0c1, s0s1).
//   U   = product of all 8 layers' gates (4x4 complex)
//   M   = U^H diag(+1,+1,-1,-1) U
//   G_jk = Re( conj(p_j) p_k M_jk ),  p = (1, -i, -i, -1)
// Runs on a single thread once per launch; libm cost here is irrelevant.
// ---------------------------------------------------------------------------
__global__ void vqc_precompute_kernel(const float* __restrict__ qw,
                                      float* __restrict__ G) {
  if (blockIdx.x != 0 || threadIdx.x != 0) return;
  float Ur[4][4] = {}, Ui[4][4] = {};
  for (int i = 0; i < 4; ++i) Ur[i][i] = 1.f;

  for (int layer = 0; layer < 8; ++layer) {
    const float* q = qw + layer * 4;
    float c, s;
    // RY on q0 (amplitude-index bit1): pairs (0,2),(1,3)
    c = cosf(0.5f * q[0]); s = sinf(0.5f * q[0]);
    for (int col = 0; col < 4; ++col)
      for (int b = 0; b < 2; ++b) {
        int j0 = b, j1 = b + 2;
        float r0 = c * Ur[j0][col] - s * Ur[j1][col];
        float i0 = c * Ui[j0][col] - s * Ui[j1][col];
        float r1 = s * Ur[j0][col] + c * Ur[j1][col];
        float i1 = s * Ui[j0][col] + c * Ui[j1][col];
        Ur[j0][col] = r0; Ui[j0][col] = i0; Ur[j1][col] = r1; Ui[j1][col] = i1;
      }
    // RZ on q0: bit1==0 -> *(c - i s), bit1==1 -> *(c + i s)
    c = cosf(0.5f * q[1]); s = sinf(0.5f * q[1]);
    for (int col = 0; col < 4; ++col)
      for (int j = 0; j < 4; ++j) {
        float sg = (j >= 2) ? s : -s;
        float r  = Ur[j][col] * c - Ui[j][col] * sg;
        float im = Ui[j][col] * c + Ur[j][col] * sg;
        Ur[j][col] = r; Ui[j][col] = im;
      }
    // RY on q1 (bit0): pairs (0,1),(2,3)
    c = cosf(0.5f * q[2]); s = sinf(0.5f * q[2]);
    for (int col = 0; col < 4; ++col)
      for (int b = 0; b < 2; ++b) {
        int j0 = 2 * b, j1 = 2 * b + 1;
        float r0 = c * Ur[j0][col] - s * Ur[j1][col];
        float i0 = c * Ui[j0][col] - s * Ui[j1][col];
        float r1 = s * Ur[j0][col] + c * Ur[j1][col];
        float i1 = s * Ui[j0][col] + c * Ui[j1][col];
        Ur[j0][col] = r0; Ui[j0][col] = i0; Ur[j1][col] = r1; Ui[j1][col] = i1;
      }
    // RZ on q1: bit0==0 -> *(c - i s), bit0==1 -> *(c + i s)
    c = cosf(0.5f * q[3]); s = sinf(0.5f * q[3]);
    for (int col = 0; col < 4; ++col)
      for (int j = 0; j < 4; ++j) {
        float sg = (j & 1) ? s : -s;
        float r  = Ur[j][col] * c - Ui[j][col] * sg;
        float im = Ui[j][col] * c + Ur[j][col] * sg;
        Ur[j][col] = r; Ui[j][col] = im;
      }
    // CNOT ctrl=q0 tgt=q1: swap rows 2,3 ; CNOT ctrl=q1 tgt=q0: swap rows 1,3
    for (int col = 0; col < 4; ++col) {
      float tr = Ur[2][col], ti = Ui[2][col];
      Ur[2][col] = Ur[3][col]; Ui[2][col] = Ui[3][col];
      Ur[3][col] = tr;         Ui[3][col] = ti;
    }
    for (int col = 0; col < 4; ++col) {
      float tr = Ur[1][col], ti = Ui[1][col];
      Ur[1][col] = Ur[3][col]; Ui[1][col] = Ui[3][col];
      Ur[3][col] = tr;         Ui[3][col] = ti;
    }
  }

  const float pr[4] = {1.f, 0.f, 0.f, -1.f};
  const float pi_[4] = {0.f, -1.f, -1.f, 0.f};
  for (int j = 0; j < 4; ++j)
    for (int k = 0; k < 4; ++k) {
      float mr = 0.f, mi = 0.f;
      for (int m = 0; m < 4; ++m) {
        float d = (m < 2) ? 1.f : -1.f;
        float ar = Ur[m][j], ai = -Ui[m][j];       // conj(U[m][j])
        float br = Ur[m][k], bi = Ui[m][k];
        mr += d * (ar * br - ai * bi);
        mi += d * (ar * bi + ai * br);
      }
      float qr = pr[j] * pr[k] + pi_[j] * pi_[k];  // conj(p_j) * p_k
      float qi = pr[j] * pi_[k] - pi_[j] * pr[k];
      G[j * 4 + k] = qr * mr - qi * mi;
    }
}

__device__ __forceinline__ float fast_tanh(float v) {
  v = fminf(fmaxf(v, -15.f), 15.f);
  float e = __expf(2.f * v);
  return (e - 1.f) / (e + 1.f);
}

// ---------------------------------------------------------------------------
// Fused policy kernel: 2 waves/block, 32 batch rows per wave (2 WMMA A-tiles,
// each loaded B-tile feeds 2 WMMAs -> half the B-operand traffic per FLOP).
// ---------------------------------------------------------------------------
__global__ __launch_bounds__(WAVES * 32)
void policy_fused_kernel(const float* __restrict__ state,
                         const float* __restrict__ b1,
                         const float* __restrict__ b2,
                         const float* __restrict__ b3,
                         const float* __restrict__ Wo1,
                         const float* __restrict__ bo1,
                         const float* __restrict__ Wo2,
                         const float* __restrict__ bo2,
                         const _Float16* __restrict__ packed,
                         const float* __restrict__ G,
                         float* __restrict__ out) {
  __shared__ _Float16 h1s[WAVES][ROWS_PER_WAVE * H1_DIM];  // 2 x 16 KB
  __shared__ _Float16 h2s[WAVES][ROWS_PER_WAVE * H2_DIM];  // 2 x 8 KB
  __shared__ float    encs[WAVES][ROWS_PER_WAVE][2];       // 512 B

  const int tid = threadIdx.x;
  const int w = tid >> 5;           // wave in block
  const int l = tid & 31;           // lane
  const int lm = l & 15;
  const int half_hi = l >> 4;       // 0: lanes 0-15, 1: lanes 16-31
  const int row_base = blockIdx.x * ROWS_PER_BLOCK + w * ROWS_PER_WAVE;

  const _Float16* pW1 = packed;
  const _Float16* pW2 = packed + PW1_HALFS;
  const _Float16* pW3 = packed + PW1_HALFS + PW2_HALFS;

  // ---------------- GEMM1: [32x64] x [64x256], A direct from global -------
  // A layout: lane holds row M = l&15; lanes 0-15: K=kk+[0..7] & kk+[16..23],
  // lanes 16-31: K=kk+[8..15] & kk+[24..31].
  const int klo = half_hi * 8;
  v16h A_[ROW_TILES][2];
  #pragma unroll
  for (int rt = 0; rt < ROW_TILES; ++rt) {
    const float* sp = state + (size_t)(row_base + rt * 16 + lm) * S_DIM;
    #pragma unroll
    for (int kk = 0; kk < 2; ++kk) {
      const float* p0 = sp + kk * 32 + klo;
      const float* p1 = p0 + 16;
      v16h a;
      #pragma unroll
      for (int i = 0; i < 8; ++i) { a[i] = (_Float16)p0[i]; a[i + 8] = (_Float16)p1[i]; }
      A_[rt][kk] = a;
    }
  }

  #pragma unroll
  for (int j = 0; j < 16; ++j) {
    v16h bt0 = *(const v16h*)(pW1 + ((size_t)(0 * 16 + j) * 512 + l * 16));
    v16h bt1 = *(const v16h*)(pW1 + ((size_t)(1 * 16 + j) * 512 + l * 16));
    const int n = j * 16 + lm;
    const float bias = b1[n];
    #pragma unroll
    for (int rt = 0; rt < ROW_TILES; ++rt) {
      v8f acc = {0.f, 0.f, 0.f, 0.f, 0.f, 0.f, 0.f, 0.f};
      acc = __builtin_amdgcn_wmma_f32_16x16x32_f16(false, A_[rt][0], false, bt0, (short)0, acc, false, false);
      acc = __builtin_amdgcn_wmma_f32_16x16x32_f16(false, A_[rt][1], false, bt1, (short)0, acc, false, false);
      // C layout: N = 16*j + lm ; M = half_hi*8 + r (within the 16-row tile)
      #pragma unroll
      for (int r = 0; r < 8; ++r) {
        float v = acc[r] + bias;
        v = v > 0.f ? v : 0.f;
        h1s[w][(rt * 16 + half_hi * 8 + r) * H1_DIM + n] = (_Float16)v;
      }
    }
  }
  __syncthreads();

  // ---------------- GEMM2: [32x256] x [256x128] ---------------------------
  v8f acc2[ROW_TILES][8];
  #pragma unroll
  for (int rt = 0; rt < ROW_TILES; ++rt)
    #pragma unroll
    for (int j = 0; j < 8; ++j)
      acc2[rt][j] = (v8f){0.f, 0.f, 0.f, 0.f, 0.f, 0.f, 0.f, 0.f};

  #pragma unroll
  for (int kk = 0; kk < 8; ++kk) {
    const int kbase = kk * 32 + klo;
    v16h Aa[ROW_TILES];
    #pragma unroll
    for (int rt = 0; rt < ROW_TILES; ++rt) {
      const _Float16* rp = &h1s[w][(rt * 16 + lm) * H1_DIM + kbase];
      v8h lo = *(const v8h*)rp;
      v8h hi = *(const v8h*)(rp + 16);
      Aa[rt] = __builtin_shufflevector(lo, hi, 0, 1, 2, 3, 4, 5, 6, 7,
                                               8, 9, 10, 11, 12, 13, 14, 15);
    }
    #pragma unroll
    for (int j = 0; j < 8; ++j) {
      v16h bt = *(const v16h*)(pW2 + ((size_t)(kk * 8 + j) * 512 + l * 16));
      #pragma unroll
      for (int rt = 0; rt < ROW_TILES; ++rt)
        acc2[rt][j] = __builtin_amdgcn_wmma_f32_16x16x32_f16(false, Aa[rt], false, bt, (short)0, acc2[rt][j], false, false);
    }
  }
  #pragma unroll
  for (int j = 0; j < 8; ++j) {
    const int n = j * 16 + lm;
    const float bias = b2[n];
    #pragma unroll
    for (int rt = 0; rt < ROW_TILES; ++rt)
      #pragma unroll
      for (int r = 0; r < 8; ++r) {
        float v = acc2[rt][j][r] + bias;
        v = v > 0.f ? v : 0.f;
        h2s[w][(rt * 16 + half_hi * 8 + r) * H2_DIM + n] = (_Float16)v;
      }
  }
  __syncthreads();

  // ---------------- GEMM3: [32x128] x [128x16(pad of 2)] ------------------
  v8f acc3[ROW_TILES];
  #pragma unroll
  for (int rt = 0; rt < ROW_TILES; ++rt)
    acc3[rt] = (v8f){0.f, 0.f, 0.f, 0.f, 0.f, 0.f, 0.f, 0.f};

  #pragma unroll
  for (int kk = 0; kk < 4; ++kk) {
    const int kbase = kk * 32 + klo;
    v16h bt = *(const v16h*)(pW3 + ((size_t)kk * 512 + l * 16));
    #pragma unroll
    for (int rt = 0; rt < ROW_TILES; ++rt) {
      const _Float16* rp = &h2s[w][(rt * 16 + lm) * H2_DIM + kbase];
      v8h lo = *(const v8h*)rp;
      v8h hi = *(const v8h*)(rp + 16);
      v16h A = __builtin_shufflevector(lo, hi, 0, 1, 2, 3, 4, 5, 6, 7,
                                               8, 9, 10, 11, 12, 13, 14, 15);
      acc3[rt] = __builtin_amdgcn_wmma_f32_16x16x32_f16(false, A, false, bt, (short)0, acc3[rt], false, false);
    }
  }
  if (lm < 2) {  // only N=0,1 are real outputs
    const float bias = b3[lm];
    #pragma unroll
    for (int rt = 0; rt < ROW_TILES; ++rt)
      #pragma unroll
      for (int r = 0; r < 8; ++r) {
        encs[w][rt * 16 + half_hi * 8 + r][lm] =
            fast_tanh(acc3[rt][r] + bias) * 3.14159265358979323846f;
      }
  }
  __syncthreads();

  // ---------------- VQC (as z = r^T G r) + output head --------------------
  // All 32 lanes active: lane l handles row l of this wave's 32 rows.
  {
    const float th0 = 0.5f * encs[w][l][0];   // |th| <= pi/2: native trig ok
    const float th1 = 0.5f * encs[w][l][1];
    const float c0 = __cosf(th0), s0 = __sinf(th0);
    const float c1 = __cosf(th1), s1 = __sinf(th1);
    const float r0 = c0 * c1, r1 = c0 * s1, r2 = s0 * c1, r3 = s0 * s1;

    const float z = G[0]  * r0 * r0 + G[5]  * r1 * r1
                  + G[10] * r2 * r2 + G[15] * r3 * r3
                  + 2.f * (G[1] * r0 * r1 + G[2]  * r0 * r2 + G[3]  * r0 * r3
                         + G[6] * r1 * r2 + G[7]  * r1 * r3 + G[11] * r2 * r3);

    const float x = (z + 1.f) * 0.5f;
    float o = bo2[0];
    #pragma unroll
    for (int i = 0; i < 16; ++i) {
      float h = fmaf(x, Wo1[i], bo1[i]);
      h = h > 0.f ? h : 0.f;
      o = fmaf(h, Wo2[i], o);
    }
    const float action = 1.f / (1.f + __expf(-o));   // ACTION_SCALE == 1.0

    const int grow = row_base + l;
    if (grow < B_TOTAL) out[grow] = action;
  }
}

// ---------------------------------------------------------------------------
extern "C" void kernel_launch(void* const* d_in, const int* in_sizes, int n_in,
                              void* d_out, int out_size, void* d_ws, size_t ws_size,
                              hipStream_t stream) {
  (void)in_sizes; (void)n_in; (void)out_size; (void)ws_size;
  const float* state = (const float*)d_in[0];
  const float* W1    = (const float*)d_in[1];
  const float* b1    = (const float*)d_in[2];
  const float* W2    = (const float*)d_in[3];
  const float* b2    = (const float*)d_in[4];
  const float* W3    = (const float*)d_in[5];
  const float* b3    = (const float*)d_in[6];
  const float* qw    = (const float*)d_in[7];
  const float* Wo1   = (const float*)d_in[8];
  const float* bo1   = (const float*)d_in[9];
  const float* Wo2   = (const float*)d_in[10];
  const float* bo2   = (const float*)d_in[11];
  float* out = (float*)d_out;
  _Float16* packed = (_Float16*)d_ws;                          // 102400 B
  float* G = (float*)((char*)d_ws + G_BYTE_OFFSET);            // 64 B

  pack_weights_kernel<<<(PACK_TOTAL + 255) / 256, 256, 0, stream>>>(W1, W2, W3, packed);
  vqc_precompute_kernel<<<1, 32, 0, stream>>>(qw, G);
  policy_fused_kernel<<<B_TOTAL / ROWS_PER_BLOCK, WAVES * 32, 0, stream>>>(
      state, b1, b2, b3, Wo1, bo1, Wo2, bo2, packed, G, out);
}
